// CrossEntropyLoss_37254546326109
// MI455X (gfx1250) — matrix operand order repaired and checked by reading, not compiled, not run on hardware
//
#include <hip/hip_runtime.h>
#include <hip/hip_bf16.h>

// CDNA5 / gfx1250: wave32, WMMA 16x16x32 f16 -> f32
typedef __attribute__((ext_vector_type(16))) _Float16 v16h;
typedef __attribute__((ext_vector_type(8)))  float    v8f;

#define IGNORE_LAB 255
#define NCLASS 21
#define DDIM   32
#define PPIX   (512*512)
#define NIMG   8
#define CHUNK  256        // pixels per WG iteration = 8 waves * 32 lanes

// per-WG partial block layout (floats)
#define PART_SUMSQ_OFF 3
#define PART_COUNT_OFF 35
#define PART_SUMS_OFF  67
#define PART_STRIDE    1091   // 3 + 32 + 32 + 32*32

#define SQ_SCALE 1048576.0f          // 2^20 fixed-point for deterministic LDS u64 atomics
#define SQ_INV   (1.0 / 1048576.0)

__global__ __launch_bounds__(256) void seg_loss_pass1(
    const float* __restrict__ logit,
    const int*   __restrict__ target,
    const float* __restrict__ feat,
    float* __restrict__ part, int G)
{
  __shared__ float fbuf[8][32][33];          // staged features, padded stride 33 (bank-conflict free)
  __shared__ int   labS[8][32];              // labels (-1 == ignore)
  __shared__ float shSums[32][32];           // per-class feature sums (classes padded to 32)
  __shared__ unsigned long long shSumsqI[32];// fixed-point sum of ||f||^2 per class
  __shared__ int   shCount[32];
  __shared__ float shRed[3 * 256];

  const int tid  = threadIdx.x;
  const int wave = tid >> 5;
  const int lane = tid & 31;
  const int half = lane >> 4;     // K-half for A/B fragments
  const int col  = lane & 15;     // row (A) / column (B,C,D) index
  const int img  = blockIdx.y;
  const int wg   = blockIdx.x;

  for (int i = tid; i < 32 * 32; i += 256) (&shSums[0][0])[i] = 0.f;
  if (tid < 32) { shSumsqI[tid] = 0ull; shCount[tid] = 0; }
  __syncthreads();

  float ceS = 0.f, varS = 0.f, intS = 0.f;
  v8f acc00 = {}, acc01 = {}, acc10 = {}, acc11 = {};   // [classTile][dimTile] f32 accumulators

  const float* logitB = logit  + (size_t)img * NCLASS * PPIX;
  const int*   tgtB   = target + (size_t)img * PPIX;
  const float* featB  = feat   + (size_t)img * PPIX * DDIM;

  const int totalChunks = PPIX / CHUNK;
  for (int ci = wg; ci < totalChunks; ci += G) {
    const int p = ci * CHUNK + wave * 32 + lane;

    const int  lab   = tgtB[p];
    const bool valid = (lab != IGNORE_LAB);
    const int  labm  = valid ? lab : -1;
    labS[wave][lane] = labm;

    // stream this lane's 32-d feature vector; stage to LDS; sum of squares in registers
    const float* fp = featB + (size_t)p * DDIM;
    float nsq = 0.f;
#pragma unroll
    for (int d = 0; d < DDIM; d += 4) {
      const float4 v = *reinterpret_cast<const float4*>(fp + d);
      fbuf[wave][lane][d + 0] = v.x;
      fbuf[wave][lane][d + 1] = v.y;
      fbuf[wave][lane][d + 2] = v.z;
      fbuf[wave][lane][d + 3] = v.w;
      nsq = fmaf(v.x, v.x, nsq);
      nsq = fmaf(v.y, v.y, nsq);
      nsq = fmaf(v.z, v.z, nsq);
      nsq = fmaf(v.w, v.w, nsq);
    }
    if (valid) {
      atomicAdd(&shSumsqI[labm], (unsigned long long)(nsq * SQ_SCALE + 0.5f));  // ds_add_u64: deterministic
      atomicAdd(&shCount[labm], 1);
    }

    // CE / VAR / Inter over the 21 class logits (kept in VGPRs via full unroll)
    float vals[NCLASS];
    float mx = -3.402823466e+38f, lsum = 0.f, llab = 0.f;
#pragma unroll
    for (int c = 0; c < NCLASS; ++c) {
      const float x = logitB[(size_t)c * PPIX + p];
      vals[c] = x;
      mx = fmaxf(mx, x);
      lsum += x;
      if (c == labm) llab = x;
    }
    float se = 0.f;
#pragma unroll
    for (int c = 0; c < NCLASS; ++c) se += __expf(vals[c] - mx);
    if (valid) {
      ceS  += __logf(se) + mx - llab;   // -log_softmax at label
      varS -= llab;
      intS += lsum - llab;
    }

    __syncthreads();  // fbuf / labS visible to whole WG

    // A fragments: one-hot(class) 16x32 f16, exact in f16.
    // 16-bit A 16x32 layout (ISA 7.12.2): lane<16 holds K {0..7,16..23}, lane>=16 holds K {8..15,24..31}
    v16h a0, a1;
#pragma unroll
    for (int j = 0; j < 16; ++j) {
      const int k  = (j < 8) ? (j + 8 * half) : (j + 8 + 8 * half);
      const int lk = labS[wave][k];
      a0[j] = (_Float16)((lk == col)      ? 1.0f : 0.0f);   // classes 0..15
      a1[j] = (_Float16)((lk == col + 16) ? 1.0f : 0.0f);   // classes 16..31 (21..31 stay zero)
    }
    // B fragments: features 32x16, split into f16 hi + lo for ~f32 accuracy.
    // 16-bit B 32x16 layout: same K order as A: lane<16 holds K {0..7,16..23}, lane>=16 holds K {8..15,24..31}
    v16h b0h, b0l, b1h, b1l;
#pragma unroll
    for (int j = 0; j < 16; ++j) {
      const int k  = (j < 8) ? (j + 8 * half) : (j + 8 + 8 * half);
      const float x0 = fbuf[wave][k][col];
      const float x1 = fbuf[wave][k][col + 16];
      const _Float16 h0 = (_Float16)x0;
      const _Float16 h1 = (_Float16)x1;
      b0h[j] = h0; b0l[j] = (_Float16)(x0 - (float)h0);
      b1h[j] = h1; b1l[j] = (_Float16)(x1 - (float)h1);
    }
    acc00 = __builtin_amdgcn_wmma_f32_16x16x32_f16(false, a0, false, b0h, (short)0, acc00, false, false);
    acc00 = __builtin_amdgcn_wmma_f32_16x16x32_f16(false, a0, false, b0l, (short)0, acc00, false, false);
    acc01 = __builtin_amdgcn_wmma_f32_16x16x32_f16(false, a0, false, b1h, (short)0, acc01, false, false);
    acc01 = __builtin_amdgcn_wmma_f32_16x16x32_f16(false, a0, false, b1l, (short)0, acc01, false, false);
    acc10 = __builtin_amdgcn_wmma_f32_16x16x32_f16(false, a1, false, b0h, (short)0, acc10, false, false);
    acc10 = __builtin_amdgcn_wmma_f32_16x16x32_f16(false, a1, false, b0l, (short)0, acc10, false, false);
    acc11 = __builtin_amdgcn_wmma_f32_16x16x32_f16(false, a1, false, b1h, (short)0, acc11, false, false);
    acc11 = __builtin_amdgcn_wmma_f32_16x16x32_f16(false, a1, false, b1l, (short)0, acc11, false, false);

    __syncthreads();  // protect fbuf/labS before next iteration overwrites
  }

  shRed[tid] = ceS; shRed[256 + tid] = varS; shRed[512 + tid] = intS;

  // Deterministic accumulator merge: waves take turns (fixed order), lanes write disjoint cells.
  // C/D 16x16 f32 layout: VGPR r -> row r (lanes 0-15) / row r+8 (lanes 16-31), column = lane%16.
  for (int w = 0; w < 8; ++w) {
    __syncthreads();
    if (wave == w) {
#pragma unroll
      for (int r = 0; r < 8; ++r) {
        const int m0 = r + 8 * half;
        shSums[m0][col]           += acc00[r];
        shSums[m0][col + 16]      += acc01[r];
        shSums[m0 + 16][col]      += acc10[r];
        shSums[m0 + 16][col + 16] += acc11[r];
      }
    }
  }
  __syncthreads();

  float* out = part + ((size_t)img * G + wg) * PART_STRIDE;
  if (tid == 0) { float s = 0.f; for (int i = 0; i < 256; ++i) s += shRed[i];        out[0] = s; }
  if (tid == 1) { float s = 0.f; for (int i = 0; i < 256; ++i) s += shRed[256 + i];  out[1] = s; }
  if (tid == 2) { float s = 0.f; for (int i = 0; i < 256; ++i) s += shRed[512 + i];  out[2] = s; }
  if (tid >= 32 && tid < 64) out[PART_SUMSQ_OFF + (tid - 32)] = (float)((double)shSumsqI[tid - 32] * SQ_INV);
  if (tid >= 64 && tid < 96) out[PART_COUNT_OFF + (tid - 64)] = (float)shCount[tid - 64];
  for (int i = tid; i < 1024; i += 256) out[PART_SUMS_OFF + i] = (&shSums[0][0])[i];
}

__global__ __launch_bounds__(256) void seg_loss_final(
    const float* __restrict__ part, int G, float* __restrict__ out)
{
  __shared__ float shC[256];
  __shared__ float shV[256];
  __shared__ float shS[NIMG * 3];

  const int tid = threadIdx.x;
  const int img = tid >> 5;     // 8 images
  const int cls = tid & 31;     // 32 padded classes -> 256 threads exactly
  const float* base = part + (size_t)img * G * PART_STRIDE;

  float cnt = 0.f, ssq = 0.f;
  float sums[DDIM];
#pragma unroll
  for (int d = 0; d < DDIM; ++d) sums[d] = 0.f;

  for (int g = 0; g < G; ++g) {
    const float* b = base + (size_t)g * PART_STRIDE;
    cnt += b[PART_COUNT_OFF + cls];
    ssq += b[PART_SUMSQ_OFF + cls];
#pragma unroll
    for (int d = 0; d < DDIM; ++d) sums[d] += b[PART_SUMS_OFF + cls * DDIM + d];
  }

  float center = 0.f;
  if (cls < NCLASS && cnt > 0.f) {
    float s2 = 0.f;
#pragma unroll
    for (int d = 0; d < DDIM; ++d) s2 = fmaf(sums[d], sums[d], s2);
    center = ssq - s2 / fmaxf(cnt, 1.f);     // sum||f||^2 - ||sum f||^2 / count
  }
  shC[tid] = center;
  shV[tid] = (cls < NCLASS) ? cnt : 0.f;
  if (cls < 3) {
    float s = 0.f;
    for (int g = 0; g < G; ++g) s += base[(size_t)g * PART_STRIDE + cls];
    shS[img * 3 + cls] = s;
  }
  __syncthreads();
  if (tid == 0) {
    float C = 0.f, V = 0.f;
    for (int i = 0; i < 256; ++i) { C += shC[i]; V += shV[i]; }
    float ce = 0.f, var = 0.f, inter = 0.f;
    for (int i = 0; i < NIMG; ++i) { ce += shS[i*3]; var += shS[i*3+1]; inter += shS[i*3+2]; }
    const float P  = (float)PPIX;
    const float CE = ce / fmaxf(V, 1.f);
    out[0] = (CE + 1.0f * (var / P) + 0.5f * (inter / P) + 0.1f * (C / P)) / (float)NIMG;
  }
}

extern "C" void kernel_launch(void* const* d_in, const int* in_sizes, int n_in,
                              void* d_out, int out_size, void* d_ws, size_t ws_size,
                              hipStream_t stream) {
  (void)in_sizes; (void)n_in; (void)out_size;
  const float* logit  = (const float*)d_in[0];
  const int*   target = (const int*)d_in[1];
  const float* feat   = (const float*)d_in[2];
  float* out  = (float*)d_out;
  float* part = (float*)d_ws;

  int G = 64;  // workgroups per image; clamp if workspace is small (deterministic for a given run)
  while (G > 1 && (size_t)NIMG * G * PART_STRIDE * sizeof(float) > ws_size) G >>= 1;

  dim3 grid(G, NIMG, 1);
  seg_loss_pass1<<<grid, 256, 0, stream>>>(logit, target, feat, part, G);
  seg_loss_final<<<1, 256, 0, stream>>>(part, G, out);
}
